// SpikingRNNRegressor_58110907515405
// MI455X (gfx1250) — matrix operand order repaired and checked by reading, not compiled, not run on hardware
//
#include <hip/hip_runtime.h>
#include <hip/hip_bf16.h>
#include <stdint.h>

// SNN regressor, persistent-kernel design for MI455X (gfx1250, wave32).
//
//  - 32 workgroups x 256 threads (8 waves). Each WG owns a 16-row batch tile
//    and iterates all T=1024 timesteps with state resident in registers/LDS.
//  - W1, W2 staged once into LDS as bf16; membrane potentials mem1/mem2 kept
//    in WMMA C-layout VGPRs (v8f per 16x16 tile, 2 tiles per wave).
//  - x tile (16x128 f32, row stride T*I) DMA'd one step ahead by the Tensor
//    Data Mover (tensor_load_to_lds, double-buffered, s_wait_tensorcnt),
//    then converted f32->bf16 cooperatively (8 elems/thread).
//  - fc1:  cur1 = x_tile(16x128 bf16 LDS) @ W1^T  : 16 tiles x 4 K-slices
//  - fc2:  cur2 = spk1(16x256 bf16 LDS)  @ W2^T  : 16 tiles x 8 K-slices
//  - readout: shfl_xor reduction + ds_add_f32, leaky integrator per batch row.

#define B_  512
#define T_  1024
#define I_  128
#define H_  256
#define BT  16          // batch tile per workgroup
#define THR 1.0f

// ---- LDS carve (dynamic shared memory, byte offsets) ----
#define OFF_W2   0u                         // [H][H] bf16   131072 B
#define OFF_W1   131072u                    // [H][I] bf16    65536 B
#define OFF_SPK  196608u                    // [BT][H] bf16    8192 B
#define OFF_XF0  204800u                    // [BT][I] f32     8192 B (buf 0)
#define OFF_XF1  212992u                    // [BT][I] f32     8192 B (buf 1)
#define OFF_XBF  221184u                    // [BT][I] bf16    4096 B
#define OFF_OACC 225280u                    // [BT] f32          64 B
#define SMEM_TOTAL 225344u

typedef __attribute__((ext_vector_type(16))) __bf16 v16bf;
typedef __attribute__((ext_vector_type(8)))  float  v8f;
typedef unsigned int u32x4 __attribute__((ext_vector_type(4)));
typedef int          i32x8 __attribute__((ext_vector_type(8)));
typedef int          i32x4 __attribute__((ext_vector_type(4)));

union V16U {
    unsigned short s[16];
    uint4          q[2];
    v16bf          v;
};

__device__ __forceinline__ unsigned short bfbits(float f) {
    union { float f; unsigned u; } a; a.f = f;
    return (unsigned short)(a.u >> 16);   // truncate f32 -> bf16
}

// pack two f32 into a dword of two bf16 (lo, hi)
__device__ __forceinline__ unsigned bfpack(float lo, float hi) {
    union { float f; unsigned u; } a, b; a.f = lo; b.f = hi;
    return (a.u >> 16) | (b.u & 0xFFFF0000u);
}

// 16 contiguous bf16 (B operand: one column, 16 consecutive K) -> v16bf
__device__ __forceinline__ v16bf ld16(const unsigned short* p) {
    V16U u;
    u.q[0] = ((const uint4*)p)[0];
    u.q[1] = ((const uint4*)p)[1];
    return u.v;
}

// A operand from LDS bf16 row: contiguous K[ks+hi*16 .. +16)
__device__ __forceinline__ v16bf ldA_lds(const unsigned short* row, int ks, int hi) {
    V16U u;
    u.q[0] = *(const uint4*)(row + ks + hi * 16);
    u.q[1] = *(const uint4*)(row + ks + hi * 16 + 8);
    return u.v;
}

__device__ __forceinline__ v8f wmma_bf16(v16bf a, v16bf b, v8f c) {
    return __builtin_amdgcn_wmma_f32_16x16x32_bf16(
        /*neg_a=*/false, a, /*neg_b=*/false, b,
        /*c_mod=*/(short)0, c, /*reuse_a=*/false, /*reuse_b=*/false);
}

// ---- Tensor Data Mover: DMA one 16x128 f32 tile of x into LDS ----
// 2D tensor: dim0 = I_ elements/row, row stride = T_*I_ elements (batch-major),
// tile = 128 x 16 rows. Groups 2/3 zero (<=2D tensor).
__device__ __forceinline__ void tdm_issue_x(const float* gaddr, unsigned lds_off) {
    unsigned long long ga = (unsigned long long)(uintptr_t)gaddr;
    u32x4 g0;
    g0.x = 1u;                                              // count=1, user D#
    g0.y = lds_off;                                         // lds_addr (bytes)
    g0.z = (unsigned)(ga & 0xFFFFFFFFu);                    // global_addr[31:0]
    g0.w = (unsigned)((ga >> 32) & 0x01FFFFFFu)             // global_addr[56:32]
         | (2u << 30);                                      // type = 2 ("image")
    i32x8 g1;
    g1[0] = (int)(2u << 16);                                // data_size = 4 B
    g1[1] = (int)((unsigned)I_ << 16);                      // tensor_dim0[15:0]
    g1[2] = (int)((((unsigned)(B_ * T_)) & 0xFFFFu) << 16); // dim0 hi | dim1 lo
    g1[3] = (int)(((unsigned)(B_ * T_)) >> 16)              // tensor_dim1[31:16]
          | (int)((unsigned)I_ << 16);                      // tile_dim0 = 128
    g1[4] = BT;                                             // tile_dim1=16, tile_dim2=0
    g1[5] = (int)(T_ * I_);                                 // tensor_dim0_stride lo
    g1[6] = 0;                                              // stride hi | dim1_stride lo
    g1[7] = 0;
    i32x4 z4 = {0, 0, 0, 0};
#if defined(__clang_major__) && (__clang_major__ >= 23)
    i32x8 z8 = {0, 0, 0, 0, 0, 0, 0, 0};
    __builtin_amdgcn_tensor_load_to_lds(g0, g1, z4, z4, z8, 0);
#else
    __builtin_amdgcn_tensor_load_to_lds(g0, g1, z4, z4, 0);
#endif
}

__global__ __launch_bounds__(256, 1)
void snn_persistent_kernel(const float* __restrict__ x,
                           const float* __restrict__ W1, const float* __restrict__ b1,
                           const float* __restrict__ W2, const float* __restrict__ b2,
                           const float* __restrict__ Wout, const float* __restrict__ bout,
                           const float* __restrict__ beta1, const float* __restrict__ beta2,
                           const float* __restrict__ betao,
                           float* __restrict__ out)
{
    extern __shared__ char smem[];
    unsigned short* w2s  = (unsigned short*)(smem + OFF_W2);
    unsigned short* w1s  = (unsigned short*)(smem + OFF_W1);
    unsigned short* spk  = (unsigned short*)(smem + OFF_SPK);
    float*          xf0  = (float*)(smem + OFF_XF0);
    float*          xf1  = (float*)(smem + OFF_XF1);
    unsigned short* xbf  = (unsigned short*)(smem + OFF_XBF);
    float*          oacc = (float*)(smem + OFF_OACC);

    const int tid  = threadIdx.x;
    const int lane = tid & 31;
    const int wv   = tid >> 5;          // wave id 0..7
    const int hi   = lane >> 4;         // half-wave selector
    const int ln   = lane & 15;         // M row (A/C) or N col (B/C)
    const int b0   = blockIdx.x * BT;

    const float* xtile_base = x + (size_t)b0 * T_ * I_;   // row b0, t=0

    // kick off the t=0 x-tile DMA before weight staging (overlap)
    if (wv == 0) tdm_issue_x(xtile_base, OFF_XF0);

    // ---- stage weights into LDS as bf16 ----
    for (int i = tid; i < H_ * I_; i += 256) w1s[i] = bfbits(W1[i]);
    for (int i = tid; i < H_ * H_; i += 256) w2s[i] = bfbits(W2[i]);
    if (tid < BT) oacc[tid] = 0.0f;

    const float bt1 = fminf(fmaxf(beta1[0], 0.0f), 1.0f);
    const float bt2 = fminf(fmaxf(beta2[0], 0.0f), 1.0f);
    const float bto = fminf(fmaxf(betao[0], 0.0f), 1.0f);
    const float bo  = bout[0];

    // this wave's two N-tiles (output feature columns)
    const int h0 = (wv * 2 + 0) * 16 + ln;
    const int h1 = (wv * 2 + 1) * 16 + ln;
    const float b1v0 = b1[h0], b1v1 = b1[h1];
    const float b2v0 = b2[h0], b2v1 = b2[h1];
    const float wo0  = Wout[h0], wo1 = Wout[h1];

    // membrane state in WMMA C-layout: reg r, lane l -> M = r + 8*hi, N = ln
    v8f mem1a = {}, mem1b = {}, mem2a = {}, mem2b = {};
    float memo = 0.0f;                       // meaningful for tid < BT

    for (int t = 0; t < T_; ++t) {
        // ---- TDM pipeline: issue t+1, wait for t, publish via barrier ----
        if (wv == 0) {
            if (t + 1 < T_) {
                tdm_issue_x(xtile_base + (size_t)(t + 1) * I_,
                            ((t + 1) & 1) ? OFF_XF1 : OFF_XF0);
                __builtin_amdgcn_s_wait_tensorcnt(1);   // tile t landed
            } else {
                __builtin_amdgcn_s_wait_tensorcnt(0);
            }
        }
        __syncthreads();

        // ---- cooperative f32 -> bf16 convert: 8 elements per thread ----
        {
            const float* xf = (t & 1) ? xf1 : xf0;
            const int base = tid * 8;                       // 2048 elems total
            float4 a = *(const float4*)(xf + base);
            float4 b = *(const float4*)(xf + base + 4);
            uint4 p;
            p.x = bfpack(a.x, a.y);
            p.y = bfpack(a.z, a.w);
            p.z = bfpack(b.x, b.y);
            p.w = bfpack(b.z, b.w);
            *(uint4*)(xbf + base) = p;
        }
        __syncthreads();

        // ---------------- fc1: cur1 = x @ W1^T ----------------
        v8f a0 = {}, a1 = {};
        const unsigned short* xrow = xbf + (size_t)ln * I_;
        #pragma unroll
        for (int ks = 0; ks < I_; ks += 32) {
            v16bf A  = ldA_lds(xrow, ks, hi);
            v16bf B0 = ld16(w1s + (size_t)h0 * I_ + ks + hi * 16);
            v16bf B1 = ld16(w1s + (size_t)h1 * I_ + ks + hi * 16);
            a0 = wmma_bf16(A, B0, a0);
            a1 = wmma_bf16(A, B1, a1);
        }

        // ---- leaky-integrate layer 1, emit spikes (bf16 0/1) to LDS ----
        #pragma unroll
        for (int r = 0; r < 8; ++r) {
            float m = mem1a[r];
            float rst = (m > THR) ? THR : 0.0f;
            m = bt1 * m + a0[r] + b1v0 - rst;
            mem1a[r] = m;
            spk[(size_t)(r + hi * 8) * H_ + h0] = (m > THR) ? (unsigned short)0x3F80 : (unsigned short)0;

            float n = mem1b[r];
            rst = (n > THR) ? THR : 0.0f;
            n = bt1 * n + a1[r] + b1v1 - rst;
            mem1b[r] = n;
            spk[(size_t)(r + hi * 8) * H_ + h1] = (n > THR) ? (unsigned short)0x3F80 : (unsigned short)0;
        }
        __syncthreads();

        // ---------------- fc2: cur2 = spk1 @ W2^T ----------------
        v8f c0 = {}, c1 = {};
        const unsigned short* arow = spk + (size_t)ln * H_;
        #pragma unroll
        for (int ks = 0; ks < H_; ks += 32) {
            v16bf A  = ldA_lds(arow, ks, hi);
            v16bf B0 = ld16(w2s + (size_t)h0 * H_ + ks + hi * 16);
            v16bf B1 = ld16(w2s + (size_t)h1 * H_ + ks + hi * 16);
            c0 = wmma_bf16(A, B0, c0);
            c1 = wmma_bf16(A, B1, c1);
        }

        // ---- leaky-integrate layer 2, spikes -> readout partials ----
        #pragma unroll
        for (int r = 0; r < 8; ++r) {
            float m = mem2a[r];
            float rst = (m > THR) ? THR : 0.0f;
            m = bt2 * m + c0[r] + b2v0 - rst;
            mem2a[r] = m;
            float p = (m > THR) ? wo0 : 0.0f;

            float n = mem2b[r];
            rst = (n > THR) ? THR : 0.0f;
            n = bt2 * n + c1[r] + b2v1 - rst;
            mem2b[r] = n;
            p += (n > THR) ? wo1 : 0.0f;

            // reduce across the 16 lanes sharing batch row M = r + 8*hi
            p += __shfl_xor(p, 1, 32);
            p += __shfl_xor(p, 2, 32);
            p += __shfl_xor(p, 4, 32);
            p += __shfl_xor(p, 8, 32);
            if (ln == 0) atomicAdd(&oacc[r + hi * 8], p);   // ds_add_f32
        }
        __syncthreads();

        // ---- output leaky integrator, one thread per batch row ----
        if (tid < BT) {
            memo = bto * memo + (oacc[tid] + bo);
            out[(size_t)(b0 + tid) * T_ + t] = memo;
            oacc[tid] = 0.0f;
        }
        __syncthreads();
    }
}

extern "C" void kernel_launch(void* const* d_in, const int* in_sizes, int n_in,
                              void* d_out, int out_size, void* d_ws, size_t ws_size,
                              hipStream_t stream) {
    (void)in_sizes; (void)n_in; (void)d_ws; (void)ws_size; (void)out_size;
    const float* x     = (const float*)d_in[0];
    const float* W1    = (const float*)d_in[1];
    const float* b1    = (const float*)d_in[2];
    const float* W2    = (const float*)d_in[3];
    const float* b2    = (const float*)d_in[4];
    const float* Wout  = (const float*)d_in[5];
    const float* bout  = (const float*)d_in[6];
    const float* beta1 = (const float*)d_in[7];
    const float* beta2 = (const float*)d_in[8];
    const float* betao = (const float*)d_in[9];
    float* out = (float*)d_out;

    hipFuncSetAttribute((const void*)snn_persistent_kernel,
                        hipFuncAttributeMaxDynamicSharedMemorySize,
                        (int)SMEM_TOTAL);

    dim3 grid(B_ / BT);   // 32 workgroups
    dim3 block(256);      // 8 waves of 32
    snn_persistent_kernel<<<grid, block, SMEM_TOTAL, stream>>>(
        x, W1, b1, W2, b2, Wout, bout, beta1, beta2, betao, out);
}
